// Grok1DecoderLayer_25469156065344
// MI455X (gfx1250) — matrix operand order repaired and checked by reading, not compiled, not run on hardware
//
#include <hip/hip_runtime.h>
#include <hip/hip_bf16.h>
#include <math.h>

// ---------------------------------------------------------------------------
// Grok-1 decoder layer for MI455X (gfx1250), wave32 + WMMA f16 (f32 accum).
// Memory-bound on fp32 weight streaming (~820MB -> ~35us @ 23.3TB/s).
// GEMMs: double-buffered LDS software pipeline; A matrices stored f16 so the
// A tile is a raw GLOBAL_LOAD_ASYNC_TO_LDS_B128 copy (ASYNCcnt) when the
// toolchain exposes the builtin; fp32 weights converted f32->f16 in registers.
// ---------------------------------------------------------------------------

typedef _Float16 v16h __attribute__((ext_vector_type(16)));
typedef _Float16 v8h  __attribute__((ext_vector_type(8)));
typedef float    v8f  __attribute__((ext_vector_type(8)));
typedef int      vi4  __attribute__((vector_size(16)));

#define kT   2048
#define kH   2048
#define kNH  16
#define kNKV 8
#define kD   128
#define kE   8
#define kI   4096
#define kQKV 4096          // (NH + 2*NKV) * D
#define kEPS 1e-5f
#define kSCALE 0.08838834764831845f   // 128^-0.5
#define kCAP 30.0f

#define BM 128
#define BN 128
#define BK 32
#define BKP 40   // padded LDS row (80B): 16B aligned, conflict-free b128 reads

#if defined(__has_builtin)
# if __has_builtin(__builtin_amdgcn_global_load_async_to_lds_b128)
#  define USE_ASYNC 1
# endif
#endif
#ifndef USE_ASYNC
# define USE_ASYNC 0
#endif

#if USE_ASYNC
typedef __attribute__((address_space(1))) vi4* gvi4p;
typedef __attribute__((address_space(3))) vi4* lvi4p;
static __device__ inline void async_copy16(const void* g, void* l) {
  __builtin_amdgcn_global_load_async_to_lds_b128(
      (gvi4p)(void*)(g), (lvi4p)(l), 0, 0);
}
static __device__ inline void wait_async0() {
# if __has_builtin(__builtin_amdgcn_s_wait_asynccnt)
  __builtin_amdgcn_s_wait_asynccnt(0);
# else
  asm volatile("s_wait_asynccnt 0" ::: "memory");
# endif
}
#endif

static __device__ inline v16h mk16(v8h lo, v8h hi) {
  v16h r;
#pragma unroll
  for (int i = 0; i < 8; ++i) { r[i] = lo[i]; r[i + 8] = hi[i]; }
  return r;
}
static __device__ inline v8h ldsv8(const _Float16* p) { return *(const v8h*)p; }

// ---------------------------------------------------------------------------
// WMMA GEMM: C[M,N] = A[M,K] * B[N,K]^T   (A f16, B fp32, C fp32)
// Optional row gather (row_idx) + dynamic row count (row_cnt) for MoE experts.
// 128x128x32 tile, 256 threads = 8 waves, each wave 32x64 (2x4 WMMA).
// Double-buffered LDS, one barrier per k-step, next tile fetched during WMMA.
// ---------------------------------------------------------------------------
__global__ __launch_bounds__(256) void gemm_wmma(
    const _Float16* __restrict__ A, int lda,
    const float* __restrict__ B, int ldb,
    float* __restrict__ C, int ldc,
    int M, int N, int K,
    const int* __restrict__ row_idx,
    const int* __restrict__ row_cnt)
{
  const int n0 = blockIdx.x * BN;
  const int m0 = blockIdx.y * BM;
  const int mcount = row_cnt ? *row_cnt : M;
  if (m0 >= mcount) return;           // uniform early-exit (before any barrier)
  (void)N;

  __shared__ alignas(16) _Float16 As[2][BM][BKP];
  __shared__ alignas(16) _Float16 Bs[2][BN][BKP];

  const int tid  = threadIdx.x;
  const int lane = tid & 31;
  const int wv   = tid >> 5;    // 0..7
  const int wm   = wv & 3;      // wave position along M (4)
  const int wn   = wv >> 2;     // wave position along N (2)
  const int l15  = lane & 15;
  const int hi16 = lane >> 4;   // 0 | 1

  // staging coords: thread covers 16 consecutive elements of one tile row
  const int sr = tid >> 1;            // 0..127 (tile row)
  const int sc = (tid & 1) * 16;      // 0 | 16

  int arow = m0 + sr;
  int ag   = row_idx ? row_idx[(arow < mcount) ? arow : 0] : arow;
  const _Float16* aptr = A + (size_t)ag * lda + sc;
  const float*    bptr = B + (size_t)(n0 + sr) * ldb + sc;

  v8f zero = {};
  v8f acc[2][4];
#pragma unroll
  for (int i = 0; i < 2; ++i)
#pragma unroll
    for (int j = 0; j < 4; ++j) acc[i][j] = zero;

  // ---- prologue: start fetch of tile 0 ----
#if USE_ASYNC
  async_copy16(aptr,     &As[0][sr][sc]);
  async_copy16(aptr + 8, &As[0][sr][sc + 8]);
#else
  uint4 areg0 = ((const uint4*)aptr)[0];
  uint4 areg1 = ((const uint4*)aptr)[1];
#endif
  float4 breg[4];
#pragma unroll
  for (int j = 0; j < 4; ++j) breg[j] = ((const float4*)bptr)[j];

  int cur = 0;
  for (int kb = 0; kb < K; kb += BK) {
    // commit staged tile into LDS[cur]
#if !USE_ASYNC
    *(uint4*)&As[cur][sr][sc]     = areg0;
    *(uint4*)&As[cur][sr][sc + 8] = areg1;
#endif
#pragma unroll
    for (int j = 0; j < 4; ++j) {
      Bs[cur][sr][sc + 4*j + 0] = (_Float16)breg[j].x;
      Bs[cur][sr][sc + 4*j + 1] = (_Float16)breg[j].y;
      Bs[cur][sr][sc + 4*j + 2] = (_Float16)breg[j].z;
      Bs[cur][sr][sc + 4*j + 3] = (_Float16)breg[j].w;
    }
#if USE_ASYNC
    wait_async0();                    // A tile resident in LDS[cur]
#endif
    __syncthreads();

    // launch fetch of next tile (overlaps with WMMA below)
    const int kn = kb + BK;
    if (kn < K) {
#if USE_ASYNC
      async_copy16(aptr + kn,     &As[cur ^ 1][sr][sc]);
      async_copy16(aptr + kn + 8, &As[cur ^ 1][sr][sc + 8]);
#else
      areg0 = ((const uint4*)(aptr + kn))[0];
      areg1 = ((const uint4*)(aptr + kn))[1];
#endif
#pragma unroll
      for (int j = 0; j < 4; ++j) breg[j] = ((const float4*)(bptr + kn))[j];
      if (kn + BK < K) __builtin_prefetch(bptr + kn + BK, 0, 1);
    }

    // compute from LDS[cur]
    v16h af[2], bf[4];
#pragma unroll
    for (int im = 0; im < 2; ++im) {
      int row = wm*32 + im*16 + l15;
      int klo = hi16 ? 8 : 0;                 // ISA 16-bit A layout
      af[im] = mk16(ldsv8(&As[cur][row][klo]), ldsv8(&As[cur][row][klo + 16]));
    }
#pragma unroll
    for (int in = 0; in < 4; ++in) {
      int n  = wn*64 + in*16 + l15;
      int ko = hi16 ? 8 : 0;                  // ISA 16-bit B layout
      bf[in] = mk16(ldsv8(&Bs[cur][n][ko]), ldsv8(&Bs[cur][n][ko + 16]));
    }
#pragma unroll
    for (int im = 0; im < 2; ++im)
#pragma unroll
      for (int in = 0; in < 4; ++in)
        acc[im][in] = __builtin_amdgcn_wmma_f32_16x16x32_f16(
            false, af[im], false, bf[in], (short)0, acc[im][in], false, false);
    cur ^= 1;
  }

#pragma unroll
  for (int im = 0; im < 2; ++im)
#pragma unroll
    for (int in = 0; in < 4; ++in)
#pragma unroll
      for (int r = 0; r < 8; ++r) {
        int rm = m0 + wm*32 + im*16 + r + hi16*8;
        int cn = n0 + wn*64 + in*16 + l15;
        if (rm < mcount) C[(size_t)rm * ldc + cn] = acc[im][in][r];
      }
}

// ---------------------------------------------------------------------------
// Flash attention: 64 q-rows per block, 4 waves (16 rows/wave), KB=32 keys.
// QK^T and PV on WMMA; online softmax with tanh logit cap; causal mask.
// GQA: head h uses kv head h/2. Output stored f16 (feeds async-A o-proj GEMM).
// ---------------------------------------------------------------------------
__global__ __launch_bounds__(128) void flash_attn(
    const float* __restrict__ qkv, _Float16* __restrict__ attn_o)
{
  const int q0   = blockIdx.x * 64;
  const int head = blockIdx.y;
  const int kvh  = head >> 1;
  const int tid  = threadIdx.x;
  const int lane = tid & 31;
  const int wv   = tid >> 5;
  const int l15  = lane & 15;
  const int hi16 = lane >> 4;
  const int hi8  = hi16 * 8;

  __shared__ alignas(16) _Float16 qs[64][128];
  __shared__ alignas(16) _Float16 ks[32][128];
  __shared__ alignas(16) _Float16 vsT[128][32];   // V transposed: [d][key]
  __shared__ alignas(16) _Float16 ps[4][16][32];  // per-wave P scratch

  { // stage Q tile (fp32 -> f16)
    int row = tid >> 1;
    int cb  = (tid & 1) * 64;
    const float* src = qkv + (size_t)(q0 + row) * kQKV + head * kD + cb;
#pragma unroll
    for (int j = 0; j < 16; ++j) {
      float4 v = ((const float4*)src)[j];
      qs[row][cb + 4*j + 0] = (_Float16)v.x;
      qs[row][cb + 4*j + 1] = (_Float16)v.y;
      qs[row][cb + 4*j + 2] = (_Float16)v.z;
      qs[row][cb + 4*j + 3] = (_Float16)v.w;
    }
  }
  __syncthreads();

  v16h qf[4];
#pragma unroll
  for (int kc = 0; kc < 4; ++kc) {
    int row = wv*16 + l15;
    int klo = kc*32 + (hi16 ? 8 : 0);
    qf[kc] = mk16(ldsv8(&qs[row][klo]), ldsv8(&qs[row][klo + 16]));
  }

  v8f zero = {};
  v8f o[8];
#pragma unroll
  for (int dt = 0; dt < 8; ++dt) o[dt] = zero;
  float m_r[8], l_r[8];
#pragma unroll
  for (int r = 0; r < 8; ++r) { m_r[r] = -1e30f; l_r[r] = 0.0f; }

  for (int kb = 0; kb < q0 + 64; kb += 32) {
    __syncthreads();
    { // stage K and V^T (fp32 -> f16)
      int kr = tid >> 2;
      int db = (tid & 3) * 32;
      const float* kp = qkv + (size_t)(kb + kr) * kQKV + kNH*kD + kvh*kD + db;
      const float* vp = qkv + (size_t)(kb + kr) * kQKV + (kNH+kNKV)*kD + kvh*kD + db;
#pragma unroll
      for (int j = 0; j < 8; ++j) {
        float4 v = ((const float4*)kp)[j];
        ks[kr][db + 4*j + 0] = (_Float16)v.x;
        ks[kr][db + 4*j + 1] = (_Float16)v.y;
        ks[kr][db + 4*j + 2] = (_Float16)v.z;
        ks[kr][db + 4*j + 3] = (_Float16)v.w;
        float4 u = ((const float4*)vp)[j];
        vsT[db + 4*j + 0][kr] = (_Float16)u.x;
        vsT[db + 4*j + 1][kr] = (_Float16)u.y;
        vsT[db + 4*j + 2][kr] = (_Float16)u.z;
        vsT[db + 4*j + 3][kr] = (_Float16)u.w;
      }
    }
    __syncthreads();

    // S = Q K^T over D=128 (4 WMMA k-chunks), two 16-key tiles
    v8f s0 = zero, s1 = zero;
#pragma unroll
    for (int kc = 0; kc < 4; ++kc) {
      int dlo = kc*32 + (hi16 ? 8 : 0);
      v16h kf0 = mk16(ldsv8(&ks[l15][dlo]),      ldsv8(&ks[l15][dlo + 16]));
      v16h kf1 = mk16(ldsv8(&ks[16 + l15][dlo]), ldsv8(&ks[16 + l15][dlo + 16]));
      s0 = __builtin_amdgcn_wmma_f32_16x16x32_f16(false, qf[kc], false, kf0, (short)0, s0, false, false);
      s1 = __builtin_amdgcn_wmma_f32_16x16x32_f16(false, qf[kc], false, kf1, (short)0, s1, false, false);
    }

    float st0[8], st1[8], alp[8];
#pragma unroll
    for (int r = 0; r < 8; ++r) {
      int qrow = q0 + wv*16 + r + hi8;
      int key0 = kb + l15;
      int key1 = kb + 16 + l15;
      float a = s0[r] * kSCALE; a = kCAP * tanhf(a * (1.0f / kCAP));
      float b = s1[r] * kSCALE; b = kCAP * tanhf(b * (1.0f / kCAP));
      if (key0 > qrow) a = -1e30f;
      if (key1 > qrow) b = -1e30f;
      float mx = fmaxf(a, b);
      mx = fmaxf(mx, __shfl_xor(mx, 1, 32));
      mx = fmaxf(mx, __shfl_xor(mx, 2, 32));
      mx = fmaxf(mx, __shfl_xor(mx, 4, 32));
      mx = fmaxf(mx, __shfl_xor(mx, 8, 32));
      float mn    = fmaxf(m_r[r], mx);
      float alpha = __expf(m_r[r] - mn);
      float p0 = __expf(a - mn);
      float p1 = __expf(b - mn);
      float ls = p0 + p1;
      ls += __shfl_xor(ls, 1, 32);
      ls += __shfl_xor(ls, 2, 32);
      ls += __shfl_xor(ls, 4, 32);
      ls += __shfl_xor(ls, 8, 32);
      l_r[r] = l_r[r] * alpha + ls;
      m_r[r] = mn;
      alp[r] = alpha;
      st0[r] = p0; st1[r] = p1;
    }
#pragma unroll
    for (int dt = 0; dt < 8; ++dt)
#pragma unroll
      for (int r = 0; r < 8; ++r) o[dt][r] *= alp[r];

    // C-layout P -> A-layout frag via wave-private LDS (LDS in-order per wave)
#pragma unroll
    for (int r = 0; r < 8; ++r) {
      ps[wv][r + hi8][l15]      = (_Float16)st0[r];
      ps[wv][r + hi8][16 + l15] = (_Float16)st1[r];
    }
    v16h pf;
    { int ko = hi16 ? 8 : 0;
      pf = mk16(ldsv8(&ps[wv][l15][ko]), ldsv8(&ps[wv][l15][ko + 16])); }

    // O += P V  (contraction over 32 keys, 8 d-tiles)
#pragma unroll
    for (int dt = 0; dt < 8; ++dt) {
      int d  = dt*16 + l15;
      int ko = hi16 ? 8 : 0;
      v16h vf = mk16(ldsv8(&vsT[d][ko]), ldsv8(&vsT[d][ko + 16]));
      o[dt] = __builtin_amdgcn_wmma_f32_16x16x32_f16(false, pf, false, vf, (short)0, o[dt], false, false);
    }
  }

#pragma unroll
  for (int dt = 0; dt < 8; ++dt)
#pragma unroll
    for (int r = 0; r < 8; ++r) {
      int t   = q0 + wv*16 + r + hi8;
      int col = head*kD + dt*16 + l15;
      attn_o[(size_t)t * (kNH * kD) + col] = (_Float16)(o[dt][r] / l_r[r]);
    }
}

// ---------------------------------------------------------------------------
// Elementwise / routing kernels
// ---------------------------------------------------------------------------
__global__ __launch_bounds__(256) void rmsnorm_f16_kernel(
    const float* __restrict__ x, const float* __restrict__ w, _Float16* __restrict__ out)
{
  int t = blockIdx.x, tid = threadIdx.x;
  const float* xr = x + (size_t)t * kH;
  float ss = 0.0f;
  for (int h = tid; h < kH; h += 256) { float v = xr[h]; ss += v * v; }
  __shared__ float red[256];
  red[tid] = ss; __syncthreads();
  for (int s = 128; s > 0; s >>= 1) { if (tid < s) red[tid] += red[tid + s]; __syncthreads(); }
  float inv = rsqrtf(red[0] * (1.0f / kH) + kEPS);
  for (int h = tid; h < kH; h += 256)
    out[(size_t)t * kH + h] = (_Float16)(xr[h] * inv * w[h]);
}

__global__ __launch_bounds__(256) void add_rmsnorm_kernel(
    const float* __restrict__ base, const float* __restrict__ y,
    const float* __restrict__ w, float* __restrict__ out)
{
  int t = blockIdx.x, tid = threadIdx.x;
  const float* yr = y + (size_t)t * kH;
  float ss = 0.0f;
  for (int h = tid; h < kH; h += 256) { float v = yr[h]; ss += v * v; }
  __shared__ float red[256];
  red[tid] = ss; __syncthreads();
  for (int s = 128; s > 0; s >>= 1) { if (tid < s) red[tid] += red[tid + s]; __syncthreads(); }
  float inv = rsqrtf(red[0] * (1.0f / kH) + kEPS);
  for (int h = tid; h < kH; h += 256)
    out[(size_t)t * kH + h] = base[(size_t)t * kH + h] + yr[h] * inv * w[h];
}

__global__ __launch_bounds__(256) void rope_kernel(
    float* __restrict__ qkv, const int* __restrict__ positions)
{
  int t = blockIdx.x;
  float pos = (float)positions[t];
  for (int i = threadIdx.x; i < 24 * 64; i += 256) {  // 16 q-heads + 8 k-heads
    int hd = i >> 6, j = i & 63;
    float* p = qkv + (size_t)t * kQKV + (hd < 16 ? hd * kD : kNH * kD + (hd - 16) * kD);
    float inv = 1.0f / __powf(10000.0f, (float)j * (1.0f / 64.0f));
    float fr = pos * inv;
    float c = cosf(fr), s = sinf(fr);
    float x1 = p[j], x2 = p[j + 64];
    p[j]      = x1 * c - x2 * s;
    p[j + 64] = x2 * c + x1 * s;
  }
}

__global__ __launch_bounds__(256) void gate_topk(
    const _Float16* __restrict__ xn, const float* __restrict__ gate_w,
    float* __restrict__ wte, int* __restrict__ counts, int* __restrict__ lists)
{
  int t = blockIdx.x, tid = threadIdx.x;
  int e = tid >> 5, lane = tid & 31;  // 8 waves, one expert each
  const _Float16* xr = xn + (size_t)t * kH;
  const float* gw = gate_w + (size_t)e * kH;
  float p = 0.0f;
  for (int h = lane; h < kH; h += 32) p += (float)xr[h] * gw[h];
#pragma unroll
  for (int m = 16; m; m >>= 1) p += __shfl_xor(p, m, 32);
  __shared__ float lg[8];
  if (lane == 0) lg[e] = p;
  __syncthreads();
  if (tid == 0) {
    float mx = lg[0];
    for (int i = 1; i < 8; ++i) mx = fmaxf(mx, lg[i]);
    float pr[8]; float s = 0.0f;
    for (int i = 0; i < 8; ++i) { pr[i] = __expf(lg[i] - mx); s += pr[i]; }
    float inv = 1.0f / s;
    int i1 = 0; for (int i = 1; i < 8; ++i) if (pr[i] > pr[i1]) i1 = i;
    int i2 = (i1 == 0) ? 1 : 0;
    for (int i = 0; i < 8; ++i) if (i != i1 && pr[i] > pr[i2]) i2 = i;
    for (int i = 0; i < 8; ++i)
      wte[(size_t)t * kE + i] = (i == i1 || i == i2) ? pr[i] * inv : 0.0f;
    int p1 = atomicAdd(&counts[i1], 1); lists[i1 * kT + p1] = t;
    int p2 = atomicAdd(&counts[i2], 1); lists[i2 * kT + p2] = t;
  }
}

__global__ __launch_bounds__(256) void gelu_mul_kernel(
    const float* __restrict__ h13, _Float16* __restrict__ act, const int* __restrict__ cnt)
{
  int m = blockIdx.y;
  if (m >= *cnt) return;
  int i = blockIdx.x * 256 + threadIdx.x;
  float g = h13[(size_t)m * (2 * kI) + i];
  float u = h13[(size_t)m * (2 * kI) + kI + i];
  act[(size_t)m * kI + i] = (_Float16)(0.5f * g * (1.0f + erff(g * 0.70710678118654752f)) * u);
}

__global__ __launch_bounds__(256) void scatter_kernel(
    const float* __restrict__ eout, const int* __restrict__ list,
    const float* __restrict__ wte, const int* __restrict__ cnt,
    int e, float* __restrict__ moe_out)
{
  int m = blockIdx.y;
  if (m >= *cnt) return;
  int tok = list[m];
  float wgt = wte[(size_t)tok * kE + e];
  int col = blockIdx.x * 256 + threadIdx.x;
  moe_out[(size_t)tok * kH + col] += wgt * eout[(size_t)m * kH + col];
}

// ---------------------------------------------------------------------------
extern "C" void kernel_launch(void* const* d_in, const int* in_sizes, int n_in,
                              void* d_out, int out_size, void* d_ws, size_t ws_size,
                              hipStream_t stream)
{
  (void)in_sizes; (void)n_in; (void)out_size; (void)ws_size;
  const float* hidden      = (const float*)d_in[0];
  const int*   positions   = (const int*)  d_in[1];
  const float* qkv_w       = (const float*)d_in[2];
  const float* o_w         = (const float*)d_in[3];
  const float* gate_w      = (const float*)d_in[4];
  const float* w13         = (const float*)d_in[5];
  const float* w2          = (const float*)d_in[6];
  const float* pre_attn_w  = (const float*)d_in[7];
  const float* post_attn_w = (const float*)d_in[8];
  const float* pre_moe_w   = (const float*)d_in[9];
  const float* post_moe_w  = (const float*)d_in[10];
  float* out = (float*)d_out;

  const size_t TH = (size_t)kT * kH;
  float* wsf = (float*)d_ws;
  _Float16* xn      = (_Float16*)wsf;                  // slot: T*H floats (f16 uses half)
  float* attn_proj  = wsf + TH;                        // T*H
  float* moe_out    = attn_proj + TH;                  // T*H
  float* regionA    = moe_out + TH;                    // T*8192 (qkv+attn_o, later h13)
  float* qkv        = regionA;                         // T*4096
  _Float16* attn_o  = (_Float16*)(regionA + (size_t)kT * kQKV);   // slot T*2048 floats
  float* h13        = regionA;                         // T*8192 (after attention done)
  _Float16* act     = (_Float16*)(regionA + (size_t)kT * (2 * kI)); // slot T*4096 floats
  float* eout       = regionA + (size_t)kT * (2 * kI) + (size_t)kT * kI; // T*2048
  float* wte        = eout + TH;                       // T*8
  int*   counts     = (int*)(wte + (size_t)kT * kE);   // 8 (+pad)
  int*   lists      = counts + 16;                     // 8*T

  // ---- attention path ----
  rmsnorm_f16_kernel<<<kT, 256, 0, stream>>>(hidden, pre_attn_w, xn);
  gemm_wmma<<<dim3(kQKV / BN, kT / BM), 256, 0, stream>>>(
      xn, kH, qkv_w, kH, qkv, kQKV, kT, kQKV, kH, nullptr, nullptr);
  rope_kernel<<<kT, 256, 0, stream>>>(qkv, positions);
  flash_attn<<<dim3(kT / 64, kNH), 128, 0, stream>>>(qkv, attn_o);
  gemm_wmma<<<dim3(kH / BN, kT / BM), 256, 0, stream>>>(
      attn_o, kH, o_w, kH, attn_proj, kH, kT, kH, kH, nullptr, nullptr);
  add_rmsnorm_kernel<<<kT, 256, 0, stream>>>(hidden, attn_proj, post_attn_w, out);

  // ---- MoE path ----
  rmsnorm_f16_kernel<<<kT, 256, 0, stream>>>(out, pre_moe_w, xn);
  (void)hipMemsetAsync(moe_out, 0, TH * sizeof(float), stream);
  (void)hipMemsetAsync(counts, 0, 16 * sizeof(int), stream);
  gate_topk<<<kT, 256, 0, stream>>>(xn, gate_w, wte, counts, lists);

  for (int e = 0; e < kE; ++e) {
    const float* w13e = w13 + (size_t)e * (2 * kI) * kH;
    const float* w2e  = w2  + (size_t)e * kH * kI;
    const int* cnt  = counts + e;
    const int* list = lists + (size_t)e * kT;
    gemm_wmma<<<dim3((2 * kI) / BN, kT / BM), 256, 0, stream>>>(
        xn, kH, w13e, kH, h13, 2 * kI, kT, 2 * kI, kH, list, cnt);
    gelu_mul_kernel<<<dim3(kI / 256, kT), 256, 0, stream>>>(h13, act, cnt);
    gemm_wmma<<<dim3(kH / BN, kT / BM), 256, 0, stream>>>(
        act, kI, w2e, kI, eout, kH, kT, kH, kI, nullptr, cnt);
    scatter_kernel<<<dim3(kH / 256, kT), 256, 0, stream>>>(eout, list, wte, cnt, e, moe_out);
  }
  add_rmsnorm_kernel<<<kT, 256, 0, stream>>>(out, moe_out, post_moe_w, out);
}
